// Context_Encoder_53712861004538
// MI455X (gfx1250) — matrix operand order repaired and checked by reading, not compiled, not run on hardware
//
#include <hip/hip_runtime.h>
#include <hip/hip_bf16.h>
#include <stdint.h>

#define D       128
#define BM      128
#define THREADS 256
#define LSTR    (D + 8)   // bf16 elements per LDS row: 272 bytes, keeps 16B alignment, rotates banks

typedef __attribute__((ext_vector_type(16))) __bf16       v16bf;
typedef __attribute__((ext_vector_type(2)))  __bf16       v2bf;
typedef __attribute__((ext_vector_type(8)))  float        v8f;
typedef __attribute__((ext_vector_type(4)))  float        v4f;
typedef __attribute__((ext_vector_type(4)))  unsigned int v4u;
typedef __attribute__((ext_vector_type(2)))  unsigned int v2u;

union FragBF {
    v4u   q[2];   // two 16B LDS loads
    v16bf v;      // WMMA operand view (32 bytes)
};

static __device__ __forceinline__ unsigned int f2bf(float f) {
    unsigned int u = __builtin_bit_cast(unsigned int, f);
    u += 0x7FFFu + ((u >> 16) & 1u);          // round-to-nearest-even
    return u >> 16;
}

// pack two f32 into one dword of bf16x2 (prefer single packed-convert VALU op)
static __device__ __forceinline__ unsigned int pack2(float x, float y) {
#if __has_builtin(__builtin_amdgcn_cvt_pk_bf16_f32)
    v2bf p = __builtin_amdgcn_cvt_pk_bf16_f32(x, y);
    return __builtin_bit_cast(unsigned int, p);
#else
    return f2bf(x) | (f2bf(y) << 16);
#endif
}

static __device__ __forceinline__ v2u pack4(v4f v) {
    v2u r;
    r.x = pack2(v.x, v.y);
    r.y = pack2(v.z, v.w);
    return r;
}

__global__ __launch_bounds__(THREADS)
void ctx_encoder_kernel(const int* __restrict__ nodes,
                        const float* __restrict__ c2e,
                        const float* __restrict__ w1,
                        const float* __restrict__ b1,
                        const float* __restrict__ w2,
                        const float* __restrict__ b2,
                        float* __restrict__ out,
                        int n_nodes)
{
    __shared__ __align__(16) unsigned short sW1[D][LSTR];
    __shared__ __align__(16) unsigned short sW2[D][LSTR];
    __shared__ __align__(16) unsigned short sX [BM][LSTR];
    __shared__ __align__(16) unsigned short sH [BM][LSTR];
    __shared__ float sB1[D];
    __shared__ float sB2[D];

    const int tid  = threadIdx.x;
    const int lane = tid & 31;
    const int wave = tid >> 5;
    const long long row0 = (long long)blockIdx.x * BM;

    // ---- weights (f32 -> bf16) into LDS: 32 float4 per row, 4096 float4 each ----
    for (int i = tid; i < D * D / 4; i += THREADS) {
        int r = i >> 5;
        int c = (i & 31) * 4;
        v4f a = *(const v4f*)(w1 + (size_t)i * 4);
        v4f b = *(const v4f*)(w2 + (size_t)i * 4);
        *(v2u*)&sW1[r][c] = pack4(a);
        *(v2u*)&sW2[r][c] = pack4(b);
    }
    if (tid < D) { sB1[tid] = b1[tid]; sB2[tid] = b2[tid]; }

    // ---- gather embedding rows (f32 -> bf16) into LDS ----
    for (int i = tid; i < BM * (D / 4); i += THREADS) {
        int r = i >> 5;
        int c = (i & 31) * 4;
        long long g = row0 + r;
        if (g >= n_nodes) g = n_nodes - 1;          // clamp (tail rows: stores guarded later)
        int node = nodes[g];
        v4f v = *(const v4f*)(c2e + (long long)node * D + c);
        *(v2u*)&sX[r][c] = pack4(v);
    }
    __syncthreads();

    // Fragment addressing (wave32, 16x16x32 bf16 WMMA):
    // A: lane L serves row (wave*16 + L%16); lanes 0-15 hold K {0-7,16-23}, lanes 16-31 hold K {8-15,24-31}
    // B: lane L serves col L%16;            lanes 0-15 hold K {0-7,16-23}, lanes 16-31 hold K {8-15,24-31}
    const int mrow = wave * 16 + (lane & 15);
    const int asel = (lane >> 4) * 8;
    const int bcol = lane & 15;
    const int bsel = (lane >> 4) * 8;
    const int crow = (lane >> 4) * 8;               // C/D: VGPR i -> row crow+i, col bcol

    // ================= layer 1: h = relu(X @ W1^T + b1) =================
    FragBF a1[4];
#pragma unroll
    for (int kc = 0; kc < 4; ++kc) {
        int kb = kc * 32 + asel;
        a1[kc].q[0] = *(const v4u*)&sX[mrow][kb];
        a1[kc].q[1] = *(const v4u*)&sX[mrow][kb + 16];
    }
#pragma unroll
    for (int nt = 0; nt < 8; ++nt) {
        float bias = sB1[nt * 16 + bcol];
        v8f acc = { bias, bias, bias, bias, bias, bias, bias, bias };
#pragma unroll
        for (int kc = 0; kc < 4; ++kc) {
            FragBF b;
            int kb = kc * 32 + bsel;
            b.q[0] = *(const v4u*)&sW1[nt * 16 + bcol][kb];
            b.q[1] = *(const v4u*)&sW1[nt * 16 + bcol][kb + 16];
            acc = __builtin_amdgcn_wmma_f32_16x16x32_bf16(
                false, a1[kc].v, false, b.v, (short)0, acc, false, false);
        }
#pragma unroll
        for (int i = 0; i < 8; ++i) {
            float h = fmaxf(acc[i], 0.0f);          // single v_max_num_f32
            sH[wave * 16 + crow + i][nt * 16 + bcol] = (unsigned short)f2bf(h);
        }
    }

    // ================= layer 2: out = relu(H @ W2^T + b2) =================
    // same-wave LDS ordering: this wave only reads the 16 rows it just wrote
    FragBF a2[4];
#pragma unroll
    for (int kc = 0; kc < 4; ++kc) {
        int kb = kc * 32 + asel;
        a2[kc].q[0] = *(const v4u*)&sH[mrow][kb];
        a2[kc].q[1] = *(const v4u*)&sH[mrow][kb + 16];
    }
    const long long orow = row0 + wave * 16 + crow;
#pragma unroll
    for (int nt = 0; nt < 8; ++nt) {
        float bias = sB2[nt * 16 + bcol];
        v8f acc = { bias, bias, bias, bias, bias, bias, bias, bias };
#pragma unroll
        for (int kc = 0; kc < 4; ++kc) {
            FragBF b;
            int kb = kc * 32 + bsel;
            b.q[0] = *(const v4u*)&sW2[nt * 16 + bcol][kb];
            b.q[1] = *(const v4u*)&sW2[nt * 16 + bcol][kb + 16];
            acc = __builtin_amdgcn_wmma_f32_16x16x32_bf16(
                false, a2[kc].v, false, b.v, (short)0, acc, false, false);
        }
#pragma unroll
        for (int i = 0; i < 8; ++i) {
            long long gr = orow + i;
            if (gr < n_nodes) {
                out[gr * D + nt * 16 + bcol] = fmaxf(acc[i], 0.0f);
            }
        }
    }
}

extern "C" void kernel_launch(void* const* d_in, const int* in_sizes, int n_in,
                              void* d_out, int out_size, void* d_ws, size_t ws_size,
                              hipStream_t stream) {
    const int*   nodes = (const int*)d_in[0];
    const float* c2e   = (const float*)d_in[1];
    const float* w1    = (const float*)d_in[2];
    const float* b1    = (const float*)d_in[3];
    const float* w2    = (const float*)d_in[4];
    const float* b2    = (const float*)d_in[5];
    float*       o     = (float*)d_out;

    int n    = in_sizes[0];
    int grid = (n + BM - 1) / BM;
    ctx_encoder_kernel<<<grid, THREADS, 0, stream>>>(nodes, c2e, w1, b1, w2, b2, o, n);
}
